// ScaledDotProductAttention_63307817943760
// MI455X (gfx1250) — matrix operand order repaired and checked by reading, not compiled, not run on hardware
//
#include <hip/hip_runtime.h>
#include <hip/hip_bf16.h>

// ---------------------------------------------------------------------------
// Fused scaled-dot-product attention for MI455X (gfx1250, wave32, WMMA).
//   energy = Q K^T / 8 ; masked (-10000) ; softmax ; out = attn @ V
// Outputs (concatenated in d_out): out [2,16,2048,64] f32, attn [2,16,2048,2048] f32
// ---------------------------------------------------------------------------

typedef __attribute__((ext_vector_type(16))) _Float16 v16h;
typedef __attribute__((ext_vector_type(4)))  _Float16 v4h;
typedef __attribute__((ext_vector_type(8)))  float    v8f;

#define SEQ     2048
#define DHEAD   64
#define NBH     32        // B*H
#define CHUNK   64        // kv rows staged per LDS round
#define KSTRIDE 80        // halves per LDS row: 160B (16B-aligned, bank-staggered)
#define PSTRIDE 40        // halves per LDS row: 80B  (16B-aligned)
#define TEMP_INV 0.125f   // 1/sqrt(64)

union FragU { v16h h; uint4 u[2]; };

__device__ __forceinline__ float4 ld4(const float* p) {
  return *reinterpret_cast<const float4*>(p);
}

// Convert 8 consecutive f32 -> f16 into dst[o..o+7]
__device__ __forceinline__ void pack8(v16h& d, int o, const float* p) {
  float4 x = ld4(p), y = ld4(p + 4);
  d[o + 0] = (_Float16)x.x; d[o + 1] = (_Float16)x.y;
  d[o + 2] = (_Float16)x.z; d[o + 3] = (_Float16)x.w;
  d[o + 4] = (_Float16)y.x; d[o + 5] = (_Float16)y.y;
  d[o + 6] = (_Float16)y.z; d[o + 7] = (_Float16)y.w;
}

// Stage 64x64 f32 tile -> LDS f16, row-major (for K: rows = kv, cols = d).
__device__ __forceinline__ void stageK(const float* __restrict__ src,
                                       _Float16* dst, int tid) {
#pragma unroll
  for (int i = 0; i < 4; ++i) {
    int idx  = i * 256 + tid;      // 0..1023 ; 16 float4-quads per row
    int row  = idx >> 4;
    int quad = idx & 15;
    float4 f = ld4(src + row * DHEAD + quad * 4);
    v4h h;
    h[0] = (_Float16)f.x; h[1] = (_Float16)f.y;
    h[2] = (_Float16)f.z; h[3] = (_Float16)f.w;
    *reinterpret_cast<v4h*>(&dst[row * KSTRIDE + quad * 4]) = h;
  }
}

// Stage 64x64 f32 tile -> LDS f16 TRANSPOSED (for V: rows = d, cols = kv).
__device__ __forceinline__ void stageVT(const float* __restrict__ src,
                                        _Float16* dst, int tid) {
#pragma unroll
  for (int i = 0; i < 4; ++i) {
    int idx  = i * 256 + tid;
    int row  = idx >> 4;           // kv
    int quad = idx & 15;           // d/4
    float4 f = ld4(src + row * DHEAD + quad * 4);
    dst[(quad * 4 + 0) * KSTRIDE + row] = (_Float16)f.x;
    dst[(quad * 4 + 1) * KSTRIDE + row] = (_Float16)f.y;
    dst[(quad * 4 + 2) * KSTRIDE + row] = (_Float16)f.z;
    dst[(quad * 4 + 3) * KSTRIDE + row] = (_Float16)f.w;
  }
}

// E(16x16) = Qtile(16x64) . K^T ; B-fragment built from LDS K rows.
// Lane holds column n = lane&15 ; k-dim (head dim) slice per ISA B layout.
__device__ __forceinline__ v8f energy_tile(const v16h& aq0, const v16h& aq1,
                                           const _Float16* Klds, int kvloc,
                                           int lane) {
  const int n      = lane & 15;
  const int dstart = (lane >> 4) * 8;    // lanes 16-31 hold k=8..15,24..31 of each frag
  const _Float16* krow = Klds + (kvloc + n) * KSTRIDE;
  FragU b0, b1;
  b0.u[0] = *reinterpret_cast<const uint4*>(krow + dstart);       // d 0..31 slice
  b0.u[1] = *reinterpret_cast<const uint4*>(krow + 16 + dstart);
  b1.u[0] = *reinterpret_cast<const uint4*>(krow + 32 + dstart);  // d 32..63 slice
  b1.u[1] = *reinterpret_cast<const uint4*>(krow + 48 + dstart);
  v8f acc = {};
  acc = __builtin_amdgcn_wmma_f32_16x16x32_f16(false, aq0, false, b0.h,
                                               (short)0, acc, false, false);
  acc = __builtin_amdgcn_wmma_f32_16x16x32_f16(false, aq1, false, b1.h,
                                               (short)0, acc, false, false);
  return acc;
}

__global__ __launch_bounds__(256)
void attn_fused(const float* __restrict__ Q, const float* __restrict__ K,
                const float* __restrict__ V, const int* __restrict__ Msk,
                float* __restrict__ Oout, float* __restrict__ Aout) {
  __shared__ __align__(16) _Float16 Klds [CHUNK * KSTRIDE];      // 10.0 KB
  __shared__ __align__(16) _Float16 VTlds[CHUNK * KSTRIDE];      // 10.0 KB
  __shared__ __align__(16) _Float16 Plds [8 * 16 * PSTRIDE];     // 10.0 KB

  const int tid  = threadIdx.x;
  const int lane = tid & 31;
  const int wave = tid >> 5;
  const int bh   = blockIdx.x >> 4;          // 0..31  (b*16 + h)
  const int qblk = blockIdx.x & 15;          // 128-row slab within the sequence
  const int b    = bh >> 4;
  const int q0   = qblk * 128 + wave * 16;   // this wave's 16 query rows

  const size_t qkv_base = (size_t)bh * SEQ * DHEAD;
  const size_t mbase    = (size_t)b * SEQ * SEQ;

  const int n    = lane & 15;                // C/B-matrix column held by lane
  const int hi   = lane >> 4;                // half-wave selector
  const int koff = hi * 8;                   // A-frag K offset for high lanes
  const int Mrow = lane & 15;                // A-frag row held by lane

  // ---- Load the wave's Q tile as two f16 A-fragments (16x32 each) ----------
  const float* qrow = Q + qkv_base + (size_t)(q0 + Mrow) * DHEAD;
  v16h aq0, aq1;
  pack8(aq0, 0, qrow + koff);        pack8(aq0, 8, qrow + 16 + koff);
  pack8(aq1, 0, qrow + 32 + koff);   pack8(aq1, 8, qrow + 48 + koff);

  // ---- Pass 1: exact online softmax statistics (m, l) per row --------------
  float m[8], l[8];
#pragma unroll
  for (int r = 0; r < 8; ++r) { m[r] = -1e30f; l[r] = 0.0f; }

  for (int c = 0; c < SEQ; c += CHUNK) {
    __syncthreads();
    stageK(K + qkv_base + (size_t)c * DHEAD, Klds, tid);
    __syncthreads();
    if (c + CHUNK < SEQ)  // warm L2/WGP$ for the next chunk
      __builtin_prefetch(K + qkv_base + (size_t)(c + CHUNK) * DHEAD + tid, 0, 0);

#pragma unroll
    for (int t = 0; t < 4; ++t) {
      const int kv = c + t * 16;
      v8f e = energy_tile(aq0, aq1, Klds, t * 16, lane);
#pragma unroll
      for (int r = 0; r < 8; ++r) {
        const int qrowg = q0 + hi * 8 + r;
        const int mv = Msk[mbase + (size_t)qrowg * SEQ + kv + n];
        float ev = (mv == 0) ? -10000.0f : e[r] * TEMP_INV;
        // row max across the 16 lanes holding this row
        float tmax = ev;
#pragma unroll
        for (int s = 1; s < 16; s <<= 1)
          tmax = fmaxf(tmax, __shfl_xor(tmax, s, 32));
        const float mn = fmaxf(m[r], tmax);
        float p = __expf(ev - mn);
        float ssum = p;
#pragma unroll
        for (int s = 1; s < 16; s <<= 1)
          ssum += __shfl_xor(ssum, s, 32);
        l[r] = l[r] * __expf(m[r] - mn) + ssum;
        m[r] = mn;
      }
    }
  }

  float rl[8];
#pragma unroll
  for (int r = 0; r < 8; ++r) rl[r] = 1.0f / l[r];   // l >= 1 always (finite mask)

  // ---- Pass 2: attention write + O = P @ V ---------------------------------
  v8f oacc[4];
#pragma unroll
  for (int j = 0; j < 4; ++j) oacc[j] = (v8f){};

  for (int c = 0; c < SEQ; c += CHUNK) {
    __syncthreads();
    stageK (K + qkv_base + (size_t)c * DHEAD, Klds, tid);
    stageVT(V + qkv_base + (size_t)c * DHEAD, VTlds, tid);
    __syncthreads();
    if (c + CHUNK < SEQ) {
      __builtin_prefetch(K + qkv_base + (size_t)(c + CHUNK) * DHEAD + tid, 0, 0);
      __builtin_prefetch(V + qkv_base + (size_t)(c + CHUNK) * DHEAD + tid, 0, 0);
    }

#pragma unroll
    for (int half = 0; half < 2; ++half) {   // 32 kv columns per P fragment
      const int kvloc = half * 32;
      _Float16* prow_w = Plds + (size_t)(wave * 16) * PSTRIDE;

#pragma unroll
      for (int t = 0; t < 2; ++t) {
        const int loc = kvloc + t * 16;
        const int kv  = c + loc;
        v8f e = energy_tile(aq0, aq1, Klds, loc, lane);
#pragma unroll
        for (int r = 0; r < 8; ++r) {
          const int qrowg = q0 + hi * 8 + r;
          const int mv = Msk[mbase + (size_t)qrowg * SEQ + kv + n];
          float ev = (mv == 0) ? -10000.0f : e[r] * TEMP_INV;
          float p  = __expf(ev - m[r]) * rl[r];
          // materialize attention (lane-contiguous along kv -> coalesced b32)
          Aout[((size_t)bh * SEQ + qrowg) * SEQ + kv + n] = p;
          // stash for A-fragment re-layout (per-wave LDS region)
          prow_w[(hi * 8 + r) * PSTRIDE + t * 16 + n] = (_Float16)p;
        }
      }

      // same-wave DS RAW: drain LDS before re-reading in A-fragment layout
      asm volatile("s_wait_dscnt 0" ::: "memory");

      const _Float16* prow = Plds + (size_t)(wave * 16 + Mrow) * PSTRIDE;
      FragU ap;
      ap.u[0] = *reinterpret_cast<const uint4*>(prow + koff);
      ap.u[1] = *reinterpret_cast<const uint4*>(prow + 16 + koff);

#pragma unroll
      for (int j = 0; j < 4; ++j) {
        // B = V[kv 32, d 16]: lane column d = j*16+n, kv slice from V^T row
        const _Float16* vrow = VTlds + (j * 16 + n) * KSTRIDE + kvloc + hi * 8;
        FragU bv;
        bv.u[0] = *reinterpret_cast<const uint4*>(vrow);
        bv.u[1] = *reinterpret_cast<const uint4*>(vrow + 16);
        oacc[j] = __builtin_amdgcn_wmma_f32_16x16x32_f16(
            false, ap.h, false, bv.h, (short)0, oacc[j], false, false);
      }
    }
  }

  // ---- Store O (f32, lane-contiguous along d) ------------------------------
#pragma unroll
  for (int j = 0; j < 4; ++j) {
#pragma unroll
    for (int r = 0; r < 8; ++r) {
      const int qrowg = q0 + hi * 8 + r;
      Oout[qkv_base + (size_t)qrowg * DHEAD + j * 16 + n] = oacc[j][r];
    }
  }
}

extern "C" void kernel_launch(void* const* d_in, const int* in_sizes, int n_in,
                              void* d_out, int out_size, void* d_ws, size_t ws_size,
                              hipStream_t stream) {
  const float* Q   = (const float*)d_in[0];
  const float* K   = (const float*)d_in[1];
  const float* V   = (const float*)d_in[2];
  const int*   Msk = (const int*)d_in[3];
  float* O = (float*)d_out;
  float* A = O + (size_t)NBH * SEQ * DHEAD;   // outputs concatenated: (out, attention)
  dim3 grid(NBH * (SEQ / 128));               // 512 blocks: one (b,h) x 128-row slab
  dim3 block(256);                            // 8 waves, one 16-row Q tile each
  attn_fused<<<grid, block, 0, stream>>>(Q, K, V, Msk, O, A);
}
